// SelfAttentionConv2d_16338055594620
// MI455X (gfx1250) — matrix operand order repaired and checked by reading, not compiled, not run on hardware
//
#include <hip/hip_runtime.h>

typedef __attribute__((ext_vector_type(16))) __bf16 v16bf;
typedef __attribute__((ext_vector_type(8)))  float  v8f;
typedef __attribute__((ext_vector_type(4)))  float  f4;

#define KH   7
#define KW   7
#define PADW 3
#define H    128
#define W    128
#define P    134            // H + 2*PAD
#define NPIX (P * P)        // 17956 padded pixels per batch
#define C    64
#define NB   2

// ---------------------------------------------------------------------------
// Kernel 1: q/k/v grouped 1x1 projections as a 64x64 block-diagonal GEMM.
//   Y[64 x NPIX] = Wblockdiag[64 x 64] * Xpad[64 x NPIX]   (x3, shared B)
// One wave = one 16(M) x 16(N) tile, K=64 via two v_wmma_f32_16x16x32_bf16.
// All loads are UNCONDITIONAL (clamped addresses + mask multiply) so the
// compiler can clause them instead of emitting per-element EXEC branches.
// Output layout: [b][pix][64] (channel-contiguous for the attention pass).
// ---------------------------------------------------------------------------
__global__ __launch_bounds__(128) void qkv_project_wmma(
    const float* __restrict__ x,
    const float* __restrict__ wq, const float* __restrict__ wk,
    const float* __restrict__ wv,
    float* __restrict__ qo, float* __restrict__ ko, float* __restrict__ vo)
{
    const int lane = threadIdx.x & 31;
    const int wave = threadIdx.x >> 5;          // M-tile: channels [16*wave, 16*wave+16)
    const int b    = blockIdx.y;
    const int l0   = blockIdx.x << 4;           // N-tile base (padded pixel index)

    const int half = lane >> 4;                 // 0: lanes 0-15, 1: lanes 16-31
    const int lmod = lane & 15;

    // ---- B fragments (Xpad tile, K=c_in), shared by q/k/v ----
    int l = l0 + lmod;
    if (l >= NPIX) l = NPIX - 1;                // clamp; stores are guarded
    const int py = l / P, px = l % P;
    const int oy = py - PADW, ox = px - PADW;
    const bool  inb   = (oy >= 0) && (oy < H) && (ox >= 0) && (ox < W);
    const float maskf = inb ? 1.0f : 0.0f;
    // clamped-valid base: any OOB lane reads channel data at pixel 0 (masked out)
    const float* xb = x + (size_t)b * C * (H * W) + (inb ? (oy * W + ox) : 0);

    float f0[16], f1[16];
    #pragma unroll
    for (int e = 0; e < 16; ++e) {              // one clause of 32 b32 loads
        const int cin = (half << 4) + e;         // B layout: K = 16*(lane/16) + e
        f0[e] = xb[(size_t)cin * (H * W)];
        f1[e] = xb[(size_t)(cin + 32) * (H * W)];
    }
    v16bf bf0, bf1;                              // K = 0..31 and 32..63
    #pragma unroll
    for (int e = 0; e < 16; ++e) {
        bf0[e] = (__bf16)(f0[e] * maskf);
        bf1[e] = (__bf16)(f1[e] * maskf);
    }

    // ---- A fragments (block-diagonal weights), K = 16*(lane/16) + e, same order as B ----
    const int m  = (wave << 4) + lmod;           // output channel (A's M = lane%16)
    const int gm = m >> 3, om = m & 7;

    auto mkA = [&](const float* __restrict__ wgt, int koff, v16bf& a) {
        #pragma unroll
        for (int e = 0; e < 16; ++e) {
            const int v  = e >> 1;
            const int kk = (half << 4) + e + koff;
            const int gk = kk >> 3, ik = kk & 7;
            // unconditional load (index always < 512); zero applied to register
            const float wval = wgt[(gm << 6) + (om << 3) + ik];
            a[e] = (__bf16)((gk == gm) ? wval : 0.0f);
        }
    };

    v16bf aq0, aq1, ak0, ak1, av0, av1;
    mkA(wq, 0, aq0);  mkA(wq, 32, aq1);
    mkA(wk, 0, ak0);  mkA(wk, 32, ak1);
    mkA(wv, 0, av0);  mkA(wv, 32, av1);

    v8f cq = {}, ck = {}, cv = {};
    cq = __builtin_amdgcn_wmma_f32_16x16x32_bf16(false, aq0, false, bf0, (short)0, cq, false, false);
    cq = __builtin_amdgcn_wmma_f32_16x16x32_bf16(false, aq1, false, bf1, (short)0, cq, false, false);
    ck = __builtin_amdgcn_wmma_f32_16x16x32_bf16(false, ak0, false, bf0, (short)0, ck, false, false);
    ck = __builtin_amdgcn_wmma_f32_16x16x32_bf16(false, ak1, false, bf1, (short)0, ck, false, false);
    cv = __builtin_amdgcn_wmma_f32_16x16x32_bf16(false, av0, false, bf0, (short)0, cv, false, false);
    cv = __builtin_amdgcn_wmma_f32_16x16x32_bf16(false, av1, false, bf1, (short)0, cv, false, false);

    // ---- store D: lane holds (M = r + 8*half, N = lmod), 8 contiguous floats ----
    const int ln = l0 + lmod;
    if (ln < NPIX) {
        const size_t rowbase = ((size_t)b * NPIX + ln) * C + (wave << 4) + (half << 3);
        *(v8f*)(qo + rowbase) = cq;
        *(v8f*)(ko + rowbase) = ck;
        *(v8f*)(vo + rowbase) = cv;
    }
}

// ---------------------------------------------------------------------------
// Kernel 2: local attention with online (flash-style) softmax.
// One thread per output pixel; q[64]/acc[64] in VGPRs as float4 vectors.
// rel is the same 8x49 pattern tiled over all 8 groups, so the rel term
// folds to 8 FMAs per window step using precomputed group-summed q.
// ---------------------------------------------------------------------------
__global__ __launch_bounds__(256) void local_attn(
    const float* __restrict__ qws, const float* __restrict__ kws,
    const float* __restrict__ vws,
    const float* __restrict__ relx, const float* __restrict__ rely,
    const float* __restrict__ bias,
    float* __restrict__ out)
{
    __shared__ float rel8[8][KH * KW];           // 1.6 KB
    for (int idx = threadIdx.x; idx < 8 * KH * KW; idx += blockDim.x) {
        const int cc = idx / (KH * KW), k = idx % (KH * KW);
        const int ky = k / KW, kx = k % KW;
        rel8[cc][k] = (cc < 4) ? relx[cc * KW + kx] : rely[(cc - 4) * KH + ky];
    }
    __syncthreads();

    const int b   = blockIdx.y;
    const int pix = blockIdx.x * blockDim.x + threadIdx.x;   // < H*W
    const int oy  = pix >> 7, ox = pix & (W - 1);
    const int lc  = (oy + PADW) * P + (ox + PADW);

    f4 q[16], acc[16];
    const f4* qrow = (const f4*)(qws + ((size_t)b * NPIX + lc) * C);
    #pragma unroll
    for (int i = 0; i < 16; ++i) {
        q[i]   = qrow[i];                        // global_load_b128
        acc[i] = (f4)(0.0f);
    }

    // qsum[cc] = sum over groups g of q[8g + cc]
    float qsum[8];
    #pragma unroll
    for (int cc = 0; cc < 8; ++cc) qsum[cc] = 0.0f;
    #pragma unroll
    for (int i = 0; i < 16; ++i) {
        const int c0 = (4 * i) & 7;              // 0 or 4
        qsum[c0 + 0] += q[i].x;
        qsum[c0 + 1] += q[i].y;
        qsum[c0 + 2] += q[i].z;
        qsum[c0 + 3] += q[i].w;
    }

    float mmax = -3.0e38f, ssum = 0.0f;
    for (int k = 0; k < KH * KW; ++k) {
        const int ky = k / KW, kx = k % KW;
        const int lk = (oy + ky) * P + (ox + kx);
        const f4* krow = (const f4*)(kws + ((size_t)b * NPIX + lk) * C);
        const f4* vrow = (const f4*)(vws + ((size_t)b * NPIX + lk) * C);

        float logit = 0.0f;
        #pragma unroll
        for (int cc = 0; cc < 8; ++cc) logit += qsum[cc] * rel8[cc][k];
        #pragma unroll
        for (int i = 0; i < 16; ++i) {
            const f4 kv = krow[i];
            logit += q[i].x * kv.x + q[i].y * kv.y + q[i].z * kv.z + q[i].w * kv.w;
        }

        const float nm    = fmaxf(mmax, logit);
        const float scale = __expf(mmax - nm);
        const float p     = __expf(logit - nm);
        ssum = ssum * scale + p;
        #pragma unroll
        for (int i = 0; i < 16; ++i) {
            const f4 vv = vrow[i];
            acc[i] = acc[i] * scale + p * vv;
        }
        mmax = nm;
    }

    const float inv = 1.0f / ssum;
    float* orow = out + (size_t)b * C * (H * W) + pix;
    #pragma unroll
    for (int i = 0; i < 16; ++i) {
        orow[(size_t)(4 * i + 0) * (H * W)] = acc[i].x * inv + bias[4 * i + 0];
        orow[(size_t)(4 * i + 1) * (H * W)] = acc[i].y * inv + bias[4 * i + 1];
        orow[(size_t)(4 * i + 2) * (H * W)] = acc[i].z * inv + bias[4 * i + 2];
        orow[(size_t)(4 * i + 3) * (H * W)] = acc[i].w * inv + bias[4 * i + 3];
    }
}

// ---------------------------------------------------------------------------
// Inputs: x, wq, wk, wv, relative_x, relative_y, bias (all float32)
// Output: (2, 64, 128, 128) float32
// Workspace: q/k/v in [b][NPIX][64] layout -> 3 * 2 * 17956 * 64 * 4 B ~ 26.3 MB
// ---------------------------------------------------------------------------
extern "C" void kernel_launch(void* const* d_in, const int* in_sizes, int n_in,
                              void* d_out, int out_size, void* d_ws, size_t ws_size,
                              hipStream_t stream)
{
    const float* x    = (const float*)d_in[0];
    const float* wq   = (const float*)d_in[1];
    const float* wk   = (const float*)d_in[2];
    const float* wv   = (const float*)d_in[3];
    const float* relx = (const float*)d_in[4];
    const float* rely = (const float*)d_in[5];
    const float* bias = (const float*)d_in[6];
    float* out = (float*)d_out;

    const size_t plane = (size_t)NB * NPIX * C;
    float* qws = (float*)d_ws;
    float* kws = qws + plane;
    float* vws = kws + plane;

    dim3 g1((NPIX + 15) / 16, NB);
    qkv_project_wmma<<<g1, 128, 0, stream>>>(x, wq, wk, wv, qws, kws, vws);

    dim3 g2((H * W) / 256, NB);
    local_attn<<<g2, 256, 0, stream>>>(qws, kws, vws, relx, rely, bias, out);
}
